// MultiHeadAttentionBlock_28295244546374
// MI455X (gfx1250) — compile-verified, matrix operand order checked by reading, not run on hardware
//
#include <hip/hip_runtime.h>
#include <hip/hip_bf16.h>

// ---------------------------------------------------------------------------
// MHA block for MI455X (gfx1250): TDM-staged bf16 WMMA GEMMs + flash attention.
// B=2, S=2048, D=1024, H=16, DK=64
// ---------------------------------------------------------------------------

#define B_   2
#define S_   2048
#define D_   1024
#define H_   16
#define DK_  64
#define M_   (B_ * S_)      // 4096 rows for the projection GEMMs

typedef __attribute__((ext_vector_type(16))) __bf16 v16bf;
typedef __attribute__((ext_vector_type(8)))  __bf16 v8bf;
typedef __attribute__((ext_vector_type(4)))  __bf16 v4bf;
typedef __attribute__((ext_vector_type(8)))  float  v8f;
typedef __attribute__((ext_vector_type(4)))  unsigned int v4u;
typedef __attribute__((ext_vector_type(8)))  int v8i;
typedef __attribute__((ext_vector_type(4)))  int v4i;

union FragU { v16bf v; v8bf h[2]; };

#if defined(__gfx1250__)

#if __has_builtin(__builtin_amdgcn_tensor_load_to_lds) && __has_builtin(__builtin_amdgcn_s_wait_tensorcnt)
#define HAVE_TDM 1
#else
#define HAVE_TDM 0
#endif

__device__ __forceinline__ v16bf load_frag(const __bf16* lo, const __bf16* hi) {
    FragU f;
    f.h[0] = *(const v8bf*)lo;
    f.h[1] = *(const v8bf*)hi;
    return f.v;
}
#define WMMA_BF16(a, b, c) \
    __builtin_amdgcn_wmma_f32_16x16x32_bf16(false, (a), false, (b), (short)0, (c), false, false)

#if HAVE_TDM
// Issue a TDM 2D tile load (tile_d1 rows x tile_d0 bf16 elems) into LDS.
// D# bit layout per CDNA5 ISA ch.8: group0 = {count, lds_addr, global_addr, type=2},
// group1 = {data_size=2B, tensor_dim0/1, tile_dim0/1, tensor_dim0_stride}.
__device__ __forceinline__ void tdm_load_2d(const __bf16* gsrc, unsigned lds_byte_addr,
                                            unsigned tensor_d0, unsigned tensor_d1,
                                            unsigned tile_d0, unsigned tile_d1,
                                            unsigned stride_d0) {
    unsigned long long ga = (unsigned long long)(uintptr_t)gsrc;
    v4u g0;
    g0[0] = 1u;                                    // count=1, user descriptor
    g0[1] = lds_byte_addr;                         // LDS dest (bytes)
    g0[2] = (unsigned)ga;                          // global_addr[31:0]
    g0[3] = (unsigned)(ga >> 32) | (2u << 30);     // global_addr[56:32] | type=2
    v8i g1;
    g1[0] = (int)(1u << 16);                       // data_size=1 (2 bytes)
    g1[1] = (int)((tensor_d0 & 0xFFFFu) << 16);    // tensor_dim0[15:0]
    g1[2] = (int)(((tensor_d0 >> 16) & 0xFFFFu) | ((tensor_d1 & 0xFFFFu) << 16));
    g1[3] = (int)(((tensor_d1 >> 16) & 0xFFFFu) | ((tile_d0 & 0xFFFFu) << 16));
    g1[4] = (int)(tile_d1 & 0xFFFFu);              // tile_dim1 (tile_dim2 = 0)
    g1[5] = (int)stride_d0;                        // tensor_dim0_stride lo32
    g1[6] = 0;
    g1[7] = 0;
    v4i z4 = {};
#if __clang_major__ >= 23
    v8i z8 = {};
    __builtin_amdgcn_tensor_load_to_lds(g0, g1, z4, z4, z8, 0);
#else
    __builtin_amdgcn_tensor_load_to_lds(g0, g1, z4, z4, 0);
#endif
}
#endif  // HAVE_TDM
#endif  // __gfx1250__

// ---------------------------------------------------------------------------
// f32 -> bf16 conversion (vectorized: 4 elems / thread)
// ---------------------------------------------------------------------------
__global__ void cvt_f32_bf16(const float* __restrict__ in, __bf16* __restrict__ out, int n) {
    int i = (blockIdx.x * blockDim.x + threadIdx.x) * 4;
    if (i < n) {
        const float4 f = *(const float4*)(in + i);
        v4bf o;
        o[0] = (__bf16)f.x; o[1] = (__bf16)f.y; o[2] = (__bf16)f.z; o[3] = (__bf16)f.w;
        *(v4bf*)(out + i) = o;
    }
}

// ---------------------------------------------------------------------------
// GEMM: Y = A[M,K] * W[N,K]^T (bf16 in, f32 accumulate).
// Block = 8 waves, 128(M) x 128(N) tile; wave tile = 32 x 64 (8 WMMAs / k-step).
// A and B 128x32 k-slices are staged in LDS, double buffered; staging is done
// by the Tensor Data Mover (wave 0 issues tensor_load_to_lds, waits TENSORcnt,
// barrier publishes) with a cooperative-copy fallback.
// MODE 0: Y bf16 -> [B,H,S,DK]   (Q/K projection)
// MODE 1: Y bf16 -> [B,H,DK,S]   (V projection, transposed for P*V)
// MODE 2: Y f32  -> [M,N]        (final output projection)
// ---------------------------------------------------------------------------
template <int MODE>
__global__ __launch_bounds__(256) void gemm_bf16(const __bf16* __restrict__ A,
                                                 const __bf16* __restrict__ W,
                                                 void* __restrict__ Yv) {
#if defined(__gfx1250__)
    __shared__ __bf16 sA[2][128 * 32];
    __shared__ __bf16 sB[2][128 * 32];

    const int lane = threadIdx.x & 31;
    const int wave = threadIdx.x >> 5;
    const int mblk = blockIdx.x * 128;
    const int nblk = blockIdx.y * 128;
    const int wm   = (wave & 3) * 32;       // wave row offset inside block tile
    const int wn   = (wave >> 2) * 64;      // wave col offset inside block tile
    const int r    = lane & 15;
    const int koff = (lane >> 4) * 8;

    const __bf16* Ag = A + (size_t)mblk * D_;
    const __bf16* Wg = W + (size_t)nblk * D_;

#if HAVE_TDM
    if (wave == 0) {
        tdm_load_2d(Ag, (unsigned)(uintptr_t)&sA[0][0], D_, 128, 32, 128, D_);
        tdm_load_2d(Wg, (unsigned)(uintptr_t)&sB[0][0], D_, 128, 32, 128, D_);
        __builtin_amdgcn_s_wait_tensorcnt(0);
    }
    __syncthreads();
#else
    {
        const int t = threadIdx.x;
        #pragma unroll
        for (int i = 0; i < 2; ++i) {
            int c = t + i * 256;            // 512 chunks of 8 elems
            *(v8bf*)(&sA[0][c * 8]) = *(const v8bf*)(Ag + (size_t)(c >> 2) * D_ + (c & 3) * 8);
            *(v8bf*)(&sB[0][c * 8]) = *(const v8bf*)(Wg + (size_t)(c >> 2) * D_ + (c & 3) * 8);
        }
    }
    __syncthreads();
#endif

    v8f acc[2][4];
    #pragma unroll
    for (int mi = 0; mi < 2; ++mi)
        #pragma unroll
        for (int nt = 0; nt < 4; ++nt) acc[mi][nt] = (v8f){};

    const int NT = D_ / 32;                 // 32 k-steps
    for (int kt = 0; kt < NT; ++kt) {
        const int cur = kt & 1;
        const int nxt = cur ^ 1;
        const int knext = (kt + 1) * 32;

#if HAVE_TDM
        if (kt + 1 < NT && wave == 0) {
            tdm_load_2d(Ag + knext, (unsigned)(uintptr_t)&sA[nxt][0], D_, 128, 32, 128, D_);
            tdm_load_2d(Wg + knext, (unsigned)(uintptr_t)&sB[nxt][0], D_, 128, 32, 128, D_);
        }
#else
        if (kt + 1 < NT) {
            const int t = threadIdx.x;
            #pragma unroll
            for (int i = 0; i < 2; ++i) {
                int c = t + i * 256;
                *(v8bf*)(&sA[nxt][c * 8]) = *(const v8bf*)(Ag + (size_t)(c >> 2) * D_ + knext + (c & 3) * 8);
                *(v8bf*)(&sB[nxt][c * 8]) = *(const v8bf*)(Wg + (size_t)(c >> 2) * D_ + knext + (c & 3) * 8);
            }
        }
#endif

        // ---- compute from current LDS tiles ----
        const __bf16* At = &sA[cur][0];
        const __bf16* Bt = &sB[cur][0];
        v16bf af0 = load_frag(At + (wm +      r) * 32 + koff, At + (wm +      r) * 32 + 16 + koff);
        v16bf af1 = load_frag(At + (wm + 16 + r) * 32 + koff, At + (wm + 16 + r) * 32 + 16 + koff);
        #pragma unroll
        for (int nt = 0; nt < 4; ++nt) {
            const int rb = wn + nt * 16 + r;
            v16bf bf = load_frag(Bt + rb * 32 + koff, Bt + rb * 32 + 16 + koff);
            acc[0][nt] = WMMA_BF16(af0, bf, acc[0][nt]);
            acc[1][nt] = WMMA_BF16(af1, bf, acc[1][nt]);
        }

#if HAVE_TDM
        if (kt + 1 < NT && wave == 0) __builtin_amdgcn_s_wait_tensorcnt(0);
#endif
        __syncthreads();   // next tiles ready + current tiles fully consumed
    }

    // ---- stores ----
    // C/D layout: VGPR g holds M = g (lanes 0-15) / g+8 (lanes 16-31), N = lane&15.
    const int c0  = lane & 15;
    const int mhi = (lane >> 4) ? 8 : 0;

    #pragma unroll
    for (int mi = 0; mi < 2; ++mi) {
        const int m0 = mblk + wm + mi * 16;
        if (MODE == 2) {
            float* Y = (float*)Yv;
            #pragma unroll
            for (int g = 0; g < 8; ++g) {
                size_t base = (size_t)(m0 + mhi + g) * D_ + nblk + wn + c0;
                Y[base +  0] = acc[mi][0][g];
                Y[base + 16] = acc[mi][1][g];
                Y[base + 32] = acc[mi][2][g];
                Y[base + 48] = acc[mi][3][g];
            }
        } else if (MODE == 0) {
            __bf16* Y = (__bf16*)Yv;
            #pragma unroll
            for (int g = 0; g < 8; ++g) {
                int m = m0 + mhi + g;
                int b = m >> 11;               // / S_
                int s = m & (S_ - 1);
                #pragma unroll
                for (int nt = 0; nt < 4; ++nt) {
                    int n  = nblk + wn + nt * 16 + c0;
                    int h  = n >> 6;           // / DK_
                    int dk = n & (DK_ - 1);
                    Y[(((size_t)(b * H_ + h)) * S_ + s) * DK_ + dk] = (__bf16)acc[mi][nt][g];
                }
            }
        } else {  // MODE 1: transposed [B,H,DK,S]; each lane owns 8 consecutive s.
            __bf16* Y = (__bf16*)Yv;
            const int b      = m0 >> 11;
            const int s_base = (m0 & (S_ - 1)) + mhi;
            #pragma unroll
            for (int nt = 0; nt < 4; ++nt) {
                int n  = nblk + wn + nt * 16 + c0;
                int h  = n >> 6;
                int dk = n & (DK_ - 1);
                v8bf pk;
                #pragma unroll
                for (int g = 0; g < 8; ++g) pk[g] = (__bf16)acc[mi][nt][g];
                *(v8bf*)(Y + ((size_t)(b * H_ + h) * DK_ + dk) * S_ + s_base) = pk;
            }
        }
    }
#endif
}

// ---------------------------------------------------------------------------
// Flash attention: one wave per 16 query rows of one (b,h).
// Qp, Kp: [B,H,S,DK] bf16;  Vt: [B,H,DK,S] bf16;  X out: [B,S,D] bf16.
// 32-key chunks: Q*K^T via 4 WMMAs, online softmax via width-16 shuffles
// (matching C-layout row striping), P via LDS round-trip to the A-fragment
// layout, then P*V via 4 WMMAs into the running O accumulator.
// ---------------------------------------------------------------------------
__global__ __launch_bounds__(128) void flash_attn(const __bf16* __restrict__ Qp,
                                                  const __bf16* __restrict__ Kp,
                                                  const __bf16* __restrict__ Vt,
                                                  __bf16* __restrict__ X) {
#if defined(__gfx1250__)
    __shared__ __bf16 plds[4][16 * 32];

    const int lane = threadIdx.x & 31;
    const int wave = threadIdx.x >> 5;
    const int idx  = blockIdx.x * 4 + wave;   // 0 .. B*H*(S/16)-1
    const int qblk = idx & 127;               // S/16 = 128
    const int bh   = idx >> 7;                // 0 .. 31
    const int b    = bh >> 4;
    const int h    = bh & 15;
    const int r    = lane & 15;
    const int koff = (lane >> 4) * 8;
    const int q0   = qblk * 16;

    const __bf16* qb = Qp + ((size_t)bh * S_ + q0 + r) * DK_;
    v16bf qf0 = load_frag(qb + koff,      qb + 16 + koff);
    v16bf qf1 = load_frag(qb + 32 + koff, qb + 48 + koff);

    v8f o0 = {}, o1 = {}, o2 = {}, o3 = {};
    v8f mrun, lrun;
    #pragma unroll
    for (int g = 0; g < 8; ++g) { mrun[g] = -3.0e38f; lrun[g] = 0.0f; }

    __bf16* pw = &plds[wave][0];
    const int mr = (lane >> 4) ? 8 : 0;
    const int c0 = lane & 15;

    for (int kb = 0; kb < S_; kb += 32) {
        const __bf16* k0 = Kp + ((size_t)bh * S_ + kb + r) * DK_;
        const __bf16* k1 = k0 + 16 * DK_;
        v16bf kb00 = load_frag(k0 + koff,      k0 + 16 + koff);
        v16bf kb01 = load_frag(k0 + 32 + koff, k0 + 48 + koff);
        v16bf kb10 = load_frag(k1 + koff,      k1 + 16 + koff);
        v16bf kb11 = load_frag(k1 + 32 + koff, k1 + 48 + koff);
        v8f s0 = {}, s1 = {};
        s0 = WMMA_BF16(qf0, kb00, s0);
        s0 = WMMA_BF16(qf1, kb01, s0);
        s1 = WMMA_BF16(qf0, kb10, s1);
        s1 = WMMA_BF16(qf1, kb11, s1);
        #pragma unroll
        for (int g = 0; g < 8; ++g) { s0[g] *= 0.125f; s1[g] *= 0.125f; }  // 1/sqrt(64)

        v8f t;
        #pragma unroll
        for (int g = 0; g < 8; ++g) t[g] = fmaxf(s0[g], s1[g]);
        #pragma unroll
        for (int off = 1; off < 16; off <<= 1) {
            #pragma unroll
            for (int g = 0; g < 8; ++g) t[g] = fmaxf(t[g], __shfl_xor(t[g], off, 16));
        }
        v8f mnew, corr, p0, p1, rs;
        #pragma unroll
        for (int g = 0; g < 8; ++g) {
            mnew[g] = fmaxf(mrun[g], t[g]);
            corr[g] = __expf(mrun[g] - mnew[g]);
            mrun[g] = mnew[g];
            p0[g]   = __expf(s0[g] - mnew[g]);
            p1[g]   = __expf(s1[g] - mnew[g]);
            rs[g]   = p0[g] + p1[g];
        }
        #pragma unroll
        for (int off = 1; off < 16; off <<= 1) {
            #pragma unroll
            for (int g = 0; g < 8; ++g) rs[g] += __shfl_xor(rs[g], off, 16);
        }
        #pragma unroll
        for (int g = 0; g < 8; ++g) {
            lrun[g] = lrun[g] * corr[g] + rs[g];
            o0[g] *= corr[g]; o1[g] *= corr[g]; o2[g] *= corr[g]; o3[g] *= corr[g];
        }

        #pragma unroll
        for (int g = 0; g < 8; ++g) {
            pw[(mr + g) * 32 + c0]      = (__bf16)p0[g];
            pw[(mr + g) * 32 + 16 + c0] = (__bf16)p1[g];
        }
        v16bf pf = load_frag(pw + r * 32 + koff, pw + r * 32 + 16 + koff);

        const __bf16* vb = Vt + ((size_t)bh * DK_ + c0) * S_ + kb;
        v16bf v0f = load_frag(vb + koff,           vb + 16 + koff);
        v16bf v1f = load_frag(vb + 16 * S_ + koff, vb + 16 * S_ + 16 + koff);
        v16bf v2f = load_frag(vb + 32 * S_ + koff, vb + 32 * S_ + 16 + koff);
        v16bf v3f = load_frag(vb + 48 * S_ + koff, vb + 48 * S_ + 16 + koff);
        o0 = WMMA_BF16(pf, v0f, o0);
        o1 = WMMA_BF16(pf, v1f, o1);
        o2 = WMMA_BF16(pf, v2f, o2);
        o3 = WMMA_BF16(pf, v3f, o3);
    }

    #pragma unroll
    for (int g = 0; g < 8; ++g) {
        float inv = 1.0f / lrun[g];
        int s = q0 + mr + g;
        size_t base = ((size_t)b * S_ + s) * D_ + h * DK_;
        X[base +  0 + c0] = (__bf16)(o0[g] * inv);
        X[base + 16 + c0] = (__bf16)(o1[g] * inv);
        X[base + 32 + c0] = (__bf16)(o2[g] * inv);
        X[base + 48 + c0] = (__bf16)(o3[g] * inv);
    }
#endif
}

// ---------------------------------------------------------------------------
// Host launcher
// ---------------------------------------------------------------------------
extern "C" void kernel_launch(void* const* d_in, const int* in_sizes, int n_in,
                              void* d_out, int out_size, void* d_ws, size_t ws_size,
                              hipStream_t stream) {
    (void)in_sizes; (void)n_in; (void)out_size; (void)ws_size;

    const float* q  = (const float*)d_in[0];
    const float* k  = (const float*)d_in[1];
    const float* v  = (const float*)d_in[2];
    const float* wq = (const float*)d_in[3];
    const float* wk = (const float*)d_in[4];
    const float* wv = (const float*)d_in[5];
    const float* wo = (const float*)d_in[6];

    const size_t NE_X = (size_t)M_ * D_;   // 4,194,304 elems
    const size_t NE_W = (size_t)D_ * D_;   // 1,048,576 elems

    char* ws = (char*)d_ws;
    size_t off = 0;
    auto alloc = [&](size_t bytes) -> void* {
        void* p = ws + off;
        off += (bytes + 255) & ~(size_t)255;
        return p;
    };

    __bf16* qb  = (__bf16*)alloc(NE_X * 2);
    __bf16* kb  = (__bf16*)alloc(NE_X * 2);
    __bf16* vb  = (__bf16*)alloc(NE_X * 2);
    __bf16* wqb = (__bf16*)alloc(NE_W * 2);
    __bf16* wkb = (__bf16*)alloc(NE_W * 2);
    __bf16* wvb = (__bf16*)alloc(NE_W * 2);
    __bf16* wob = (__bf16*)alloc(NE_W * 2);
    __bf16* Qp  = (__bf16*)alloc(NE_X * 2);   // [B,H,S,DK]
    __bf16* Kp  = (__bf16*)alloc(NE_X * 2);   // [B,H,S,DK]
    __bf16* Vt  = (__bf16*)alloc(NE_X * 2);   // [B,H,DK,S]
    __bf16* Xb  = (__bf16*)alloc(NE_X * 2);   // [B,S,D]

    const int cvt_threads = 256;
    auto cvt = [&](const float* in, __bf16* out, size_t n) {
        int blocks = (int)((n / 4 + cvt_threads - 1) / cvt_threads);
        cvt_f32_bf16<<<blocks, cvt_threads, 0, stream>>>(in, out, (int)n);
    };
    cvt(q,  qb,  NE_X);
    cvt(k,  kb,  NE_X);
    cvt(v,  vb,  NE_X);
    cvt(wq, wqb, NE_W);
    cvt(wk, wkb, NE_W);
    cvt(wv, wvb, NE_W);
    cvt(wo, wob, NE_W);

    dim3 ggrid(M_ / 128, D_ / 128);   // (32, 8)
    gemm_bf16<0><<<ggrid, 256, 0, stream>>>(qb, wqb, (void*)Qp);
    gemm_bf16<0><<<ggrid, 256, 0, stream>>>(kb, wkb, (void*)Kp);
    gemm_bf16<1><<<ggrid, 256, 0, stream>>>(vb, wvb, (void*)Vt);

    dim3 fgrid((B_ * H_ * (S_ / 16)) / 4);  // 1024 blocks * 4 waves
    flash_attn<<<fgrid, 128, 0, stream>>>(Qp, Kp, Vt, Xb);

    gemm_bf16<2><<<ggrid, 256, 0, stream>>>(Xb, wob, d_out);
}
